// MyDense_49624052138671
// MI455X (gfx1250) — 8-bit matrix operand order repaired and checked by reading, compiled, not run on hardware
//
#include <hip/hip_runtime.h>
#include <stdint.h>


typedef __attribute__((ext_vector_type(16))) int          v16i;
typedef __attribute__((ext_vector_type(8)))  float        v8f;
typedef __attribute__((ext_vector_type(4)))  unsigned int u32x4;
typedef __attribute__((ext_vector_type(8)))  int          i32x8;
typedef __attribute__((ext_vector_type(4)))  int          i32x4;

#if defined(__has_builtin)
#  if __has_builtin(__builtin_amdgcn_tensor_load_to_lds)
#    define USE_TDM 1
#  endif
#endif
#ifndef USE_TDM
#  define USE_TDM 0
#endif

__device__ __forceinline__ unsigned cvt_e4m3(float f) {
  float c = fminf(fmaxf(f, -448.0f), 448.0f);
  unsigned u = __float_as_uint(c);
  unsigned sign = (u >> 24) & 0x80u;
  unsigned a = u & 0x7FFFFFFFu;
  if (a < 0x3C800000u) {
    unsigned m = (unsigned)(int)rintf(__uint_as_float(a) * 512.0f);
    return sign | m;
  }
  unsigned r = a + 0x7FFFFu + ((a >> 20) & 1u);
  unsigned e = (r >> 23) - 127u + 7u;
  unsigned m = (r >> 20) & 7u;
  return sign | (e << 3) | m;
}

__device__ __forceinline__ void block_amax_atomic(float local, unsigned* dst) {
#pragma unroll
  for (int off = 16; off > 0; off >>= 1)
    local = fmaxf(local, __shfl_xor(local, off, 32));
  __shared__ float wred[8];
  int lane = threadIdx.x & 31, wid = threadIdx.x >> 5;
  if (lane == 0) wred[wid] = local;
  __syncthreads();
  if (threadIdx.x == 0) {
    float m = wred[0];
    int nw = (int)(blockDim.x >> 5);
    for (int i = 1; i < nw; ++i) m = fmaxf(m, wred[i]);
    atomicMax(dst, __float_as_uint(m));
  }
}

__global__ void k_quant_x(const float* __restrict__ x, uint8_t* __restrict__ xq,
                          unsigned* __restrict__ amax,
                          const float* __restrict__ scale, int n4) {
  const float s = scale[0];
  float local = 0.0f;
  int stride = (int)(gridDim.x * blockDim.x);
  for (int i = (int)(blockIdx.x * blockDim.x + threadIdx.x); i < n4; i += stride) {
    float4 v = ((const float4*)x)[i];
    local = fmaxf(local, fmaxf(fmaxf(fabsf(v.x), fabsf(v.y)),
                               fmaxf(fabsf(v.z), fabsf(v.w))));
    unsigned p = cvt_e4m3(v.x * s) | (cvt_e4m3(v.y * s) << 8) |
                 (cvt_e4m3(v.z * s) << 16) | (cvt_e4m3(v.w * s) << 24);
    ((unsigned*)xq)[i] = p;
  }
  block_amax_atomic(local, amax);
}

__global__ void k_quant_w(const float* __restrict__ w, uint8_t* __restrict__ wqT,
                          unsigned* __restrict__ amax,
                          const float* __restrict__ scale, int K, int N) {
  const float s = scale[1];
  __shared__ uint8_t tile[64 * 68];
  int kt = blockIdx.x * 64, nt = blockIdx.y * 64;
  float local = 0.0f;
#pragma unroll
  for (int i = 0; i < 16; ++i) {
    int lin = (int)threadIdx.x + i * 256;
    int kr = lin >> 6, nc = lin & 63;
    float v = w[(size_t)(kt + kr) * N + nt + nc];
    local = fmaxf(local, fabsf(v));
    tile[nc * 68 + kr] = (uint8_t)cvt_e4m3(v * s);
  }
  __syncthreads();
#pragma unroll
  for (int i = 0; i < 4; ++i) {
    int lin = (int)threadIdx.x + i * 256;
    int nr = lin >> 4, kd = lin & 15;
    *(unsigned*)(wqT + (size_t)(nt + nr) * K + kt + kd * 4) =
        *(const unsigned*)(tile + nr * 68 + kd * 4);
  }
  block_amax_atomic(local, amax + 1);
}

constexpr int BM = 128, BN = 128, BK = 128;
constexpr int LDT = 144;
constexpr int TB  = 128 * LDT;

#if USE_TDM
__device__ __forceinline__ void tdm_load2d(void* lds_ptr, const void* gptr,
                                           unsigned tensor_w, unsigned tensor_h,
                                           unsigned row_stride) {
  unsigned long long ga = (unsigned long long)(uintptr_t)gptr;
  unsigned lds = (unsigned)(uintptr_t)lds_ptr;
  u32x4 g0;
  g0[0] = 1u;
  g0[1] = lds;
  g0[2] = (unsigned)ga;
  g0[3] = (unsigned)(ga >> 32) | (2u << 30);
  i32x8 g1;
  g1[0] = (int)((1u << 20) | (4u << 22) | (3u << 25));
  g1[1] = (int)((tensor_w & 0xFFFFu) << 16);
  g1[2] = (int)((tensor_w >> 16) | ((tensor_h & 0xFFFFu) << 16));
  g1[3] = (int)((tensor_h >> 16) | (128u << 16));
  g1[4] = (int)128u;
  g1[5] = (int)row_stride;
  g1[6] = 0;
  g1[7] = 0;
  i32x4 z4 = {0, 0, 0, 0};
#if __clang_major__ >= 23
  i32x8 z8 = {0, 0, 0, 0, 0, 0, 0, 0};
  __builtin_amdgcn_tensor_load_to_lds(g0, g1, z4, z4, z8, 0);
#else
  __builtin_amdgcn_tensor_load_to_lds(g0, g1, z4, z4, 0);
#endif
}
#endif

__global__ void __launch_bounds__(256)
k_gemm(const uint8_t* __restrict__ xq, const uint8_t* __restrict__ wqT,
       float* __restrict__ out, const float* __restrict__ scale,
       int M, int N, int K) {
  __shared__ __align__(16) uint8_t smem[2 * 2 * TB];

  const int tid  = (int)threadIdx.x;
  const int lane = tid & 31;
  const int wid  = tid >> 5;
  const int wm   = wid & 1;
  const int wn   = wid >> 1;
  const int lr   = lane & 15;
  const int hb   = lane >> 4;
  const int m0   = (int)blockIdx.y * BM;
  const int n0   = (int)blockIdx.x * BN;
  const int KT   = K / BK;
  const float s  = 1.0f / (scale[0] * scale[1]);

  v8f acc[4][2];
#pragma unroll
  for (int mt = 0; mt < 4; ++mt)
#pragma unroll
    for (int nt = 0; nt < 2; ++nt) acc[mt][nt] = (v8f){0,0,0,0,0,0,0,0};

#if USE_TDM
  if (wid == 0) {
    tdm_load2d(smem + 0 * TB, xq  + (size_t)m0 * K, (unsigned)K, (unsigned)M, (unsigned)K);
    tdm_load2d(smem + 1 * TB, wqT + (size_t)n0 * K, (unsigned)K, (unsigned)N, (unsigned)K);
  }
#endif

  for (int t = 0; t < KT; ++t) {
    const int b = t & 1;
    uint8_t* Asb = smem + (size_t)(2 * b + 0) * TB;
    uint8_t* Bsb = smem + (size_t)(2 * b + 1) * TB;

#if USE_TDM
    if (wid == 0) {
      if (t + 1 < KT) {
        int nb = (t + 1) & 1;
        tdm_load2d(smem + (size_t)(2 * nb + 0) * TB,
                   xq  + (size_t)m0 * K + (size_t)(t + 1) * BK,
                   (unsigned)K, (unsigned)M, (unsigned)K);
        tdm_load2d(smem + (size_t)(2 * nb + 1) * TB,
                   wqT + (size_t)n0 * K + (size_t)(t + 1) * BK,
                   (unsigned)K, (unsigned)N, (unsigned)K);
        __builtin_amdgcn_s_wait_tensorcnt(2);
      } else {
        __builtin_amdgcn_s_wait_tensorcnt(0);
      }
    }
    __syncthreads();
#else
    {
      const uint8_t* ga = xq  + (size_t)m0 * K + (size_t)t * BK;
      const uint8_t* gb = wqT + (size_t)n0 * K + (size_t)t * BK;
#pragma unroll
      for (int i = 0; i < 4; ++i) {
        int lin = tid + i * 256;
        int row = lin >> 3, col = (lin & 7) * 16;
        *(uint4*)(Asb + row * LDT + col) = *(const uint4*)(ga + (size_t)row * K + col);
        *(uint4*)(Bsb + row * LDT + col) = *(const uint4*)(gb + (size_t)row * K + col);
      }
    }
    __syncthreads();
#endif

    union AFu { v16i v; unsigned long long u[8]; } af[4];
    union BFu { v16i v; uint4 q[4]; } bf[2];
#pragma unroll
    for (int mt = 0; mt < 4; ++mt) {
      const uint8_t* p = Asb + (wm * 64 + mt * 16 + lr) * LDT + hb * 16;
#pragma unroll
      for (int c = 0; c < 8; ++c)
        af[mt].u[c] = *(const unsigned long long*)(p + (c >> 1) * 32 + (c & 1) * 8);
    }
#pragma unroll
    for (int nt = 0; nt < 2; ++nt) {
      const uint8_t* p = Bsb + (wn * 32 + nt * 16 + lr) * LDT + hb * 16;
#pragma unroll
      for (int g = 0; g < 4; ++g)
        bf[nt].q[g] = *(const uint4*)(p + g * 32);
    }
#pragma unroll
    for (int mt = 0; mt < 4; ++mt)
#pragma unroll
      for (int nt = 0; nt < 2; ++nt)
        acc[mt][nt] = __builtin_amdgcn_wmma_f32_16x16x128_fp8_fp8(
            af[mt].v, bf[nt].v, (short)0, acc[mt][nt], false, false);

    __syncthreads();
  }

#pragma unroll
  for (int mt = 0; mt < 4; ++mt)
#pragma unroll
    for (int nt = 0; nt < 2; ++nt) {
      int mbase = m0 + wm * 64 + mt * 16 + hb * 8;
      int n     = n0 + wn * 32 + nt * 16 + lr;
#pragma unroll
      for (int j = 0; j < 8; ++j)
        out[(size_t)(mbase + j) * N + n] = acc[mt][nt][j] * s;
    }
}

__global__ void k_finalize(const unsigned* __restrict__ amax_bits,
                           const float* __restrict__ scale,
                           float* __restrict__ tail) {
  int i = (int)threadIdx.x;
  if (i < 2) {
    float amax = __uint_as_float(amax_bits[i]);
    float sc = scale[i];
    float e = floorf(log2f(448.0f / amax));
    float sf = roundf(exp2f(fabsf(e)));
    sf = (amax > 0.0f) ? sf : sc;
    sf = isinf(amax) ? sf : sc;
    sf = (e < 0.0f) ? (1.0f / sf) : sf;
    tail[i] = sf;
    tail[2 + i] = 0.0f;
  }
}

extern "C" void kernel_launch(void* const* d_in, const int* in_sizes, int n_in,
                              void* d_out, int out_size, void* d_ws, size_t ws_size,
                              hipStream_t stream) {
  (void)n_in; (void)out_size; (void)ws_size;
  const float* x     = (const float*)d_in[0];
  const float* w     = (const float*)d_in[1];
  const float* scale = (const float*)d_in[2];
  const int K = 2048, N = 2048;
  const int M = in_sizes[0] / K;

  uint8_t*  xq   = (uint8_t*)d_ws;
  uint8_t*  wqT  = xq + (size_t)M * K;
  unsigned* amax = (unsigned*)(wqT + (size_t)N * K);

  hipMemsetAsync(amax, 0, 2 * sizeof(unsigned), stream);
  k_quant_x<<<2048, 256, 0, stream>>>(x, xq, amax, scale, M * K / 4);
  dim3 gw(K / 64, N / 64);
  k_quant_w<<<gw, 256, 0, stream>>>(w, wqT, amax, scale, K, N);
  dim3 gg(N / BN, M / BM);
  k_gemm<<<gg, 256, 0, stream>>>(xq, wqT, (float*)d_out, scale, M, N, K);
  k_finalize<<<1, 32, 0, stream>>>(amax, scale, (float*)d_out + (size_t)M * N);
}
